// BaseVAE_609885356034
// MI455X (gfx1250) — matrix operand order repaired and checked by reading, not compiled, not run on hardware
//
#include <hip/hip_runtime.h>

// ---------------------------------------------------------------------------
// CDNA5 (gfx1250) VAE forward: PointNet encoder + BN(train) + reparam +
// decoder + fused chamfer-L1.  Matrix work uses v_wmma_f32_16x16x32_bf16.
// Activations/weights are pre-packed to bf16 pairs so the GEMM inner loop is
// pure b128 loads + WMMA.
// ---------------------------------------------------------------------------

typedef __attribute__((ext_vector_type(16))) __bf16   v16bf;
typedef __attribute__((ext_vector_type(8)))  float    v8f;
typedef __attribute__((ext_vector_type(8)))  unsigned v8u;

#define NB   8
#define NPTS 6144
#define NTOK (NB * NPTS)      // 49152
#define NB_CHUNK 2048

static __device__ __forceinline__ unsigned pack_bf16(float a, float b) {
  unsigned lo = __builtin_bit_cast(unsigned short, (__bf16)a);
  unsigned hi = __builtin_bit_cast(unsigned short, (__bf16)b);
  return lo | (hi << 16);
}

// ---------------------------------------------------------------------------
// conv1: 3 -> 128 channels (K too small for WMMA; VALU). Raw out + BN stats.
__global__ void __launch_bounds__(128) k_conv1(
    const float* __restrict__ x,      // (B,3,N) channels-first
    const float* __restrict__ w,      // (128,3)
    const float* __restrict__ bias,   // (128)
    float* __restrict__ out,          // (NTOK,128)
    float* __restrict__ sum, float* __restrict__ sq)
{
  int o  = threadIdx.x;
  int t0 = blockIdx.x * 64;
  float w0 = w[o*3+0], w1 = w[o*3+1], w2 = w[o*3+2], bb = bias[o];
  float s = 0.f, q = 0.f;
  for (int i = 0; i < 64; ++i) {
    int t = t0 + i;
    int b = t / NPTS, n = t % NPTS;
    float x0 = x[(b*3+0)*NPTS + n];
    float x1 = x[(b*3+1)*NPTS + n];
    float x2 = x[(b*3+2)*NPTS + n];
    float v  = fmaf(w2, x2, fmaf(w1, x1, fmaf(w0, x0, bb)));
    out[(size_t)t*128 + o] = v;
    s += v; q += v*v;
  }
  atomicAdd(&sum[o], s);
  atomicAdd(&sq[o],  q);
}

// ---------------------------------------------------------------------------
// BN finalize: (sum,sumsq,count,gamma,beta) -> per-channel scale/shift
__global__ void k_bnfin(const float* __restrict__ sum, const float* __restrict__ sq,
                        const float* __restrict__ g,   const float* __restrict__ be,
                        float* __restrict__ scale, float* __restrict__ shift,
                        int C, float invCount)
{
  int c = blockIdx.x*blockDim.x + threadIdx.x;
  if (c >= C) return;
  float mu  = sum[c] * invCount;
  float var = sq[c]  * invCount - mu*mu;
  float sc  = g[c] * rsqrtf(var + 1e-5f);
  scale[c] = sc;
  shift[c] = be[c] - mu*sc;
}

// ---------------------------------------------------------------------------
// Apply BN + ReLU once, pack f32 activations -> bf16 pairs ([M][K/2] dwords).
__global__ void __launch_bounds__(256) k_pack_act(
    const float2* __restrict__ h, const float* __restrict__ scale,
    const float* __restrict__ shift, unsigned* __restrict__ outp, int K2)
{
  size_t idx = (size_t)blockIdx.x * 256 + threadIdx.x;
  int k2 = (int)(idx % (size_t)K2);
  float2 hv = h[idx];
  int k = 2 * k2;
  float a0 = fmaxf(fmaf(hv.x, scale[k],   shift[k]),   0.f);
  float a1 = fmaxf(fmaf(hv.y, scale[k+1], shift[k+1]), 0.f);
  outp[idx] = pack_bf16(a0, a1);
}

// Pack weights f32 -> bf16 pairs (no BN).
__global__ void __launch_bounds__(256) k_pack_w(
    const float2* __restrict__ w, unsigned* __restrict__ outp, int total)
{
  int idx = blockIdx.x * 256 + threadIdx.x;
  if (idx >= total) return;
  float2 v = w[idx];
  outp[idx] = pack_bf16(v.x, v.y);
}

// ---------------------------------------------------------------------------
// GEMM on pre-packed bf16: out[M][O] = act[M][K] * w^T, raw f32 out + BN
// statistics.  Inner loop: 4x b128 loads + 1 wmma per 32-deep K step.
__global__ void __launch_bounds__(256) k_gemm_bn(
    const unsigned* __restrict__ abf,   // [M][K/2] bf16 pairs
    const unsigned* __restrict__ wbf,   // [O][K/2] bf16 pairs
    const float* __restrict__ bias,     // [O]
    float* __restrict__ out,            // [M][O] raw
    float* __restrict__ sumOut, float* __restrict__ sqOut,
    int K, int O)
{
  int lane = threadIdx.x & 31;
  int wave = threadIdx.x >> 5;
  bool hi  = lane >= 16;
  int lm   = lane & 15;
  int m0   = (blockIdx.x * 8 + wave) * 16;
  int n    = blockIdx.y * 16 + lm;
  int K2   = K >> 1;

  const unsigned* arow = abf + (size_t)(m0 + lm) * K2;
  const unsigned* wrow = wbf + (size_t)n * K2;

  v8f acc = {};
  for (int k2 = 0; k2 < K2; k2 += 16) {
    // A fragment dwords: v<4 -> k2 + (hi?4:0) + v ; v>=4 -> k2+8+(hi?4:0)+v-4
    uint4 a0 = *(const uint4*)(arow + k2 + (hi ? 4 : 0));
    uint4 a1 = *(const uint4*)(arow + k2 + 8 + (hi ? 4 : 0));
    // B fragment dwords: same as A (v<4 -> k2+(hi?4:0)+v ; v>=4 -> k2+8+(hi?4:0)+v-4)
    uint4 b0 = *(const uint4*)(wrow + k2 + (hi ? 4 : 0));
    uint4 b1 = *(const uint4*)(wrow + k2 + 8 + (hi ? 4 : 0));
    v8u au, bu;
    au[0]=a0.x; au[1]=a0.y; au[2]=a0.z; au[3]=a0.w;
    au[4]=a1.x; au[5]=a1.y; au[6]=a1.z; au[7]=a1.w;
    bu[0]=b0.x; bu[1]=b0.y; bu[2]=b0.z; bu[3]=b0.w;
    bu[4]=b1.x; bu[5]=b1.y; bu[6]=b1.z; bu[7]=b1.w;
    v16bf af    = __builtin_bit_cast(v16bf, au);
    v16bf bfrag = __builtin_bit_cast(v16bf, bu);
    acc = __builtin_amdgcn_wmma_f32_16x16x32_bf16(
        false, af, false, bfrag, (short)0, acc, false, false);
  }

  float bn_ = bias[n];
  float s = 0.f, q = 0.f;
  #pragma unroll
  for (int r = 0; r < 8; ++r) {
    int m = m0 + r + (hi ? 8 : 0);
    float val = acc[r] + bn_;
    out[(size_t)m * O + n] = val;
    s += val; q += val*val;
  }
  s += __shfl_down(s, 16, 32);
  q += __shfl_down(q, 16, 32);
  if (!hi) { atomicAdd(&sumOut[n], s); atomicAdd(&sqOut[n], q); }
}

// ---------------------------------------------------------------------------
// Global max pool over points with BN4 applied on the fly: pooled (B,512)
__global__ void __launch_bounds__(128) k_pool(
    const float* __restrict__ h4, const float* __restrict__ sc,
    const float* __restrict__ sh, float* __restrict__ pooled)
{
  int c = blockIdx.y * 128 + threadIdx.x;
  int b = blockIdx.x;
  float s = sc[c], t = sh[c];
  float mx = -3.4e38f;
  const float* p = h4 + (size_t)b * NPTS * 512 + c;
  for (int n = 0; n < NPTS; ++n)
    mx = fmaxf(mx, fmaf(p[(size_t)n * 512], s, t));
  pooled[b*512 + c] = mx;
}

// ---------------------------------------------------------------------------
// Small FC (8 batch rows): optional fused input BN / ReLU, optional BN stats.
__global__ void __launch_bounds__(64) k_fc(
    const float* __restrict__ in,   const float* __restrict__ scIn,
    const float* __restrict__ shIn, int reluIn,
    const float* __restrict__ w,    const float* __restrict__ bias,
    float* __restrict__ out, float* __restrict__ sum, float* __restrict__ sq,
    int K, int O)
{
  int o = blockIdx.x * 64 + threadIdx.x;
  if (o >= O) return;
  const float* wr = w + (size_t)o * K;
  float s = 0.f, q = 0.f;
  for (int b = 0; b < 8; ++b) {
    const float* ir = in + (size_t)b * K;
    float acc = bias[o];
    for (int k = 0; k < K; ++k) {
      float v = ir[k];
      if (scIn)  v = fmaf(v, scIn[k], shIn[k]);
      if (reluIn) v = fmaxf(v, 0.f);
      acc = fmaf(v, wr[k], acc);
    }
    out[(size_t)b * O + o] = acc;
    s += acc; q += acc*acc;
  }
  if (sum) { sum[o] = s; sq[o] = q; }
}

// ---------------------------------------------------------------------------
// Reparameterization + KL: z = m + sqrt(softplus(vraw)+1e-8)*eps
__global__ void __launch_bounds__(128) k_reparam(
    const float* __restrict__ mr, const float* __restrict__ vr,
    const float* __restrict__ eps, float* __restrict__ z, float* __restrict__ kl)
{
  __shared__ float red[128];
  int j = threadIdx.x;
  float acc = 0.f;
  for (int b = 0; b < 8; ++b) {
    float m  = mr[b*128 + j];
    float xr = vr[b*128 + j];
    float sp = (xr > 20.f) ? xr : log1pf(__expf(xr));
    float v  = sp + 1e-8f;
    z[b*128 + j] = fmaf(sqrtf(v), eps[b*128 + j], m);
    acc += 0.5f * (-__logf(v) + v + m*m - 1.f);
  }
  red[j] = acc; __syncthreads();
  for (int st = 64; st > 0; st >>= 1) {
    if (j < st) red[j] += red[j + st];
    __syncthreads();
  }
  if (j == 0) kl[0] = red[0] * 0.125f;
}

// ---------------------------------------------------------------------------
// Fused chamfer row-min: min_j d2(a_i, b_j) with b staged in LDS chunks and
// 16x16 dot-product tiles via WMMA (K=32, only k=0..2 populated).
__global__ void __launch_bounds__(256) k_chamfer_min(
    const float* __restrict__ A, int aChanMajor,
    const float* __restrict__ Bp, int bChanMajor,
    float* __restrict__ dmin)
{
  __shared__ uint2 sbp[NB_CHUNK];
  __shared__ float sb2[NB_CHUNK];

  int lane = threadIdx.x & 31, wave = threadIdx.x >> 5;
  bool hi = lane >= 16;
  int lm  = lane & 15;
  int b   = blockIdx.y;
  int arow0 = blockIdx.x * 128 + wave * 16;

  float ax, ay, az;
  {
    int r = arow0 + lm;
    if (aChanMajor) {
      ax = A[(b*3+0)*NPTS + r]; ay = A[(b*3+1)*NPTS + r]; az = A[(b*3+2)*NPTS + r];
    } else {
      const float* p = A + ((size_t)b * NPTS + r) * 3;
      ax = p[0]; ay = p[1]; az = p[2];
    }
  }
  __bf16 hx = (__bf16)ax, hy = (__bf16)ay, hz = (__bf16)az;
  float a2lane = (float)hx*(float)hx + (float)hy*(float)hy + (float)hz*(float)hz;
  v16bf af = {};
  if (!hi) { af[0] = hx; af[1] = hy; af[2] = hz; }

  float a2m[8];
  #pragma unroll
  for (int r = 0; r < 8; ++r)
    a2m[r] = __shfl(a2lane, r + (hi ? 8 : 0), 32);

  float rm[8];
  #pragma unroll
  for (int r = 0; r < 8; ++r) rm[r] = 3.4e38f;

  for (int c0 = 0; c0 < NPTS; c0 += NB_CHUNK) {
    for (int i = threadIdx.x; i < NB_CHUNK; i += 256) {
      int jg = c0 + i;
      float bx, by, bz;
      if (bChanMajor) {
        bx = Bp[(b*3+0)*NPTS + jg]; by = Bp[(b*3+1)*NPTS + jg]; bz = Bp[(b*3+2)*NPTS + jg];
      } else {
        const float* p = Bp + ((size_t)b * NPTS + jg) * 3;
        bx = p[0]; by = p[1]; bz = p[2];
      }
      __bf16 bhx = (__bf16)bx, bhy = (__bf16)by, bhz = (__bf16)bz;
      sbp[i] = make_uint2(pack_bf16(bx, by),
                          (unsigned)__builtin_bit_cast(unsigned short, bhz));
      sb2[i] = (float)bhx*(float)bhx + (float)bhy*(float)bhy + (float)bhz*(float)bhz;
    }
    __syncthreads();

    for (int n0 = 0; n0 < NB_CHUNK; n0 += 16) {
      v16bf bfrag = {};
      if (!hi) {
        uint2 pp = sbp[n0 + lm];
        bfrag[0] = __builtin_bit_cast(__bf16, (unsigned short)(pp.x & 0xffffu));
        bfrag[1] = __builtin_bit_cast(__bf16, (unsigned short)(pp.x >> 16));
        bfrag[2] = __builtin_bit_cast(__bf16, (unsigned short)(pp.y & 0xffffu));
      }
      v8f acc = {};
      acc = __builtin_amdgcn_wmma_f32_16x16x32_bf16(
          false, af, false, bfrag, (short)0, acc, false, false);
      float b2n = sb2[n0 + lm];
      #pragma unroll
      for (int r = 0; r < 8; ++r) {
        float d2 = fmaxf(a2m[r] + b2n - 2.f * acc[r], 0.f);
        rm[r] = fminf(rm[r], d2);
      }
    }
    __syncthreads();
  }

  #pragma unroll
  for (int r = 0; r < 8; ++r) {
    float v = rm[r];
    v = fminf(v, __shfl_xor(v, 1, 32));
    v = fminf(v, __shfl_xor(v, 2, 32));
    v = fminf(v, __shfl_xor(v, 4, 32));
    v = fminf(v, __shfl_xor(v, 8, 32));
    rm[r] = v;
  }
  if (lm == 0) {
    int mbase = arow0 + (hi ? 8 : 0);
    #pragma unroll
    for (int r = 0; r < 8; ++r)
      dmin[(size_t)b * NPTS + mbase + r] = rm[r];
  }
}

// ---------------------------------------------------------------------------
// Final reduce: recon = (0.5/(8*N)) * sum(sqrt(dmin_both)); write outputs.
__global__ void __launch_bounds__(256) k_final(
    const float* __restrict__ dmin2, const float* __restrict__ kl,
    float* __restrict__ out, int total)
{
  __shared__ float red[256];
  float acc = 0.f;
  for (int i = threadIdx.x; i < total; i += 256) acc += sqrtf(dmin2[i]);
  red[threadIdx.x] = acc; __syncthreads();
  for (int st = 128; st > 0; st >>= 1) {
    if (threadIdx.x < st) red[threadIdx.x] += red[threadIdx.x + st];
    __syncthreads();
  }
  if (threadIdx.x == 0) {
    float recon = red[0] * (0.5f / (8.f * (float)NPTS));
    float k = kl[0];
    out[0] = recon + k;
    out[1] = k;
    out[2] = recon;
  }
}

// ---------------------------------------------------------------------------
// Workspace layout (4-byte units)
#define OFF_BUFA   ((size_t)0)                       // 49152*256 f32
#define OFF_BUFB   (OFF_BUFA + (size_t)NTOK*256)     // 49152*512 f32
#define OFF_POOL   (OFF_BUFB + (size_t)NTOK*512)     // 4096
#define OFF_STATS  (OFF_POOL + 4096)                 // 3584 (zeroed each call)
#define OFF_SCALE  (OFF_STATS + 3584)                // 3584
#define OFF_MRAW   (OFF_SCALE + 3584)
#define OFF_VRAW   (OFF_MRAW + 1024)
#define OFF_Z      (OFF_VRAW + 1024)
#define OFF_T1M    (OFF_Z + 1024)
#define OFF_T2M    (OFF_T1M + 2048)
#define OFF_T1V    (OFF_T2M + 1024)
#define OFF_T2V    (OFF_T1V + 2048)
#define OFF_D1     (OFF_T2V + 1024)
#define OFF_D2     (OFF_D1 + 2048)
#define OFF_DMIN   (OFF_D2 + 2048)                   // 2 * 8 * 6144
#define OFF_KL     (OFF_DMIN + (size_t)2*NB*NPTS)    // 16
#define OFF_BFC    (OFF_KL + 16)                     // NTOK*128 dwords (bf16 of h1/h3)
#define OFF_BFD    (OFF_BFC + (size_t)NTOK*128)      // NTOK*64 dwords (bf16 of h2)
#define OFF_WB2    (OFF_BFD + (size_t)NTOK*64)       // 128*64
#define OFF_WB3    (OFF_WB2 + 8192)                  // 256*64
#define OFF_WB4    (OFF_WB3 + 16384)                 // 512*128

extern "C" void kernel_launch(void* const* d_in, const int* in_sizes, int n_in,
                              void* d_out, int out_size, void* d_ws, size_t ws_size,
                              hipStream_t stream) {
  (void)in_sizes; (void)n_in; (void)out_size; (void)ws_size;
  auto P = [&](int i) { return (const float*)d_in[i]; };
  const float* x   = P(0);
  const float* eps = P(1);
  const float *c1w = P(2),  *c1b = P(3),  *c1g = P(4),  *c1be = P(5);
  const float *c2w = P(6),  *c2b = P(7),  *c2g = P(8),  *c2be = P(9);
  const float *c3w = P(10), *c3b = P(11), *c3g = P(12), *c3be = P(13);
  const float *c4w = P(14), *c4b = P(15), *c4g = P(16), *c4be = P(17);
  const float *f1mw = P(18), *f1mb = P(19), *b1mg = P(20), *b1mb = P(21);
  const float *f2mw = P(22), *f2mb = P(23), *b2mg = P(24), *b2mb = P(25);
  const float *f3mw = P(26), *f3mb = P(27);
  const float *f1vw = P(28), *f1vb = P(29), *b1vg = P(30), *b1vb = P(31);
  const float *f2vw = P(32), *f2vb = P(33), *b2vg = P(34), *b2vb = P(35);
  const float *f3vw = P(36), *f3vb = P(37);
  const float *d1w = P(38), *d1b = P(39), *d2w = P(40), *d2b = P(41);
  const float *dfw = P(42), *dfb = P(43);

  float* ws  = (float*)d_ws;
  float* out = (float*)d_out;
  float* h1  = ws + OFF_BUFA;   // [NTOK][128] raw
  float* h2  = ws + OFF_BUFB;   // [NTOK][128] raw
  float* h3  = ws + OFF_BUFA;   // [NTOK][256] raw (h1 dead)
  float* h4  = ws + OFF_BUFB;   // [NTOK][512] raw (h2 dead)
  float* pooled = ws + OFF_POOL;
  float* st  = ws + OFF_STATS;
  float* sc  = ws + OFF_SCALE;
  float *s1=st+0,   *s2=st+256,  *s3=st+512,  *s4=st+1024;
  float *sf1m=st+2048, *sf2m=st+2560, *sf1v=st+2816, *sf2v=st+3328;
  float *x1=sc+0,   *x2=sc+256,  *x3=sc+512,  *x4=sc+1024;
  float *xf1m=sc+2048, *xf2m=sc+2560, *xf1v=sc+2816, *xf2v=sc+3328;
  float* mraw = ws + OFF_MRAW; float* vraw = ws + OFF_VRAW; float* z = ws + OFF_Z;
  float* t1m = ws + OFF_T1M; float* t2m = ws + OFF_T2M;
  float* t1v = ws + OFF_T1V; float* t2v = ws + OFF_T2V;
  float* dd1 = ws + OFF_D1;  float* dd2 = ws + OFF_D2;
  float* dmin = ws + OFF_DMIN; float* kl = ws + OFF_KL;
  unsigned* bfC = (unsigned*)(ws + OFF_BFC);   // bf16 of h1 (K=128) / h3 (K=256)
  unsigned* bfD = (unsigned*)(ws + OFF_BFD);   // bf16 of h2 (K=128)
  unsigned* wb2 = (unsigned*)(ws + OFF_WB2);
  unsigned* wb3 = (unsigned*)(ws + OFF_WB3);
  unsigned* wb4 = (unsigned*)(ws + OFF_WB4);
  float* y = out + 3;                          // (8, 6144, 3)

  const float invTok = 1.0f / (float)NTOK, invB = 0.125f;

  hipMemsetAsync(st, 0, 3584 * sizeof(float), stream);

  // pack weights to bf16 (independent of everything else)
  k_pack_w<<<32,  256, 0, stream>>>((const float2*)c2w, wb2, 128*64);
  k_pack_w<<<64,  256, 0, stream>>>((const float2*)c3w, wb3, 256*64);
  k_pack_w<<<256, 256, 0, stream>>>((const float2*)c4w, wb4, 512*128);

  // ---- encoder convs ----
  k_conv1<<<NTOK/64, 128, 0, stream>>>(x, c1w, c1b, h1, s1, s1+128);
  k_bnfin<<<1, 128, 0, stream>>>(s1, s1+128, c1g, c1be, x1, x1+128, 128, invTok);
  k_pack_act<<<NTOK*64/256, 256, 0, stream>>>((const float2*)h1, x1, x1+128, bfC, 64);
  k_gemm_bn<<<dim3(NTOK/128, 8),  256, 0, stream>>>(bfC, wb2, c2b, h2, s2, s2+128, 128, 128);
  k_bnfin<<<1, 128, 0, stream>>>(s2, s2+128, c2g, c2be, x2, x2+128, 128, invTok);
  k_pack_act<<<NTOK*64/256, 256, 0, stream>>>((const float2*)h2, x2, x2+128, bfD, 64);
  k_gemm_bn<<<dim3(NTOK/128, 16), 256, 0, stream>>>(bfD, wb3, c3b, h3, s3, s3+256, 128, 256);
  k_bnfin<<<1, 256, 0, stream>>>(s3, s3+256, c3g, c3be, x3, x3+256, 256, invTok);
  k_pack_act<<<NTOK*128/256, 256, 0, stream>>>((const float2*)h3, x3, x3+256, bfC, 128);
  k_gemm_bn<<<dim3(NTOK/128, 32), 256, 0, stream>>>(bfC, wb4, c4b, h4, s4, s4+512, 256, 512);
  k_bnfin<<<2, 256, 0, stream>>>(s4, s4+512, c4g, c4be, x4, x4+512, 512, invTok);
  k_pool<<<dim3(8, 4), 128, 0, stream>>>(h4, x4, x4+512, pooled);

  // ---- mean head ----
  k_fc<<<4, 64, 0, stream>>>(pooled, nullptr, nullptr, 0, f1mw, f1mb, t1m, sf1m, sf1m+256, 512, 256);
  k_bnfin<<<1, 256, 0, stream>>>(sf1m, sf1m+256, b1mg, b1mb, xf1m, xf1m+256, 256, invB);
  k_fc<<<2, 64, 0, stream>>>(t1m, xf1m, xf1m+256, 1, f2mw, f2mb, t2m, sf2m, sf2m+128, 256, 128);
  k_bnfin<<<1, 128, 0, stream>>>(sf2m, sf2m+128, b2mg, b2mb, xf2m, xf2m+128, 128, invB);
  k_fc<<<2, 64, 0, stream>>>(t2m, xf2m, xf2m+128, 1, f3mw, f3mb, mraw, nullptr, nullptr, 128, 128);
  // ---- var head ----
  k_fc<<<4, 64, 0, stream>>>(pooled, nullptr, nullptr, 0, f1vw, f1vb, t1v, sf1v, sf1v+256, 512, 256);
  k_bnfin<<<1, 256, 0, stream>>>(sf1v, sf1v+256, b1vg, b1vb, xf1v, xf1v+256, 256, invB);
  k_fc<<<2, 64, 0, stream>>>(t1v, xf1v, xf1v+256, 1, f2vw, f2vb, t2v, sf2v, sf2v+128, 256, 128);
  k_bnfin<<<1, 128, 0, stream>>>(sf2v, sf2v+128, b2vg, b2vb, xf2v, xf2v+128, 128, invB);
  k_fc<<<2, 64, 0, stream>>>(t2v, xf2v, xf2v+128, 1, f3vw, f3vb, vraw, nullptr, nullptr, 128, 128);

  // ---- reparameterization + KL ----
  k_reparam<<<1, 128, 0, stream>>>(mraw, vraw, eps, z, kl);

  // ---- decoder (y -> d_out+3) ----
  k_fc<<<4,   64, 0, stream>>>(z,   nullptr, nullptr, 0, d1w, d1b, dd1, nullptr, nullptr, 128, 256);
  k_fc<<<4,   64, 0, stream>>>(dd1, nullptr, nullptr, 1, d2w, d2b, dd2, nullptr, nullptr, 256, 256);
  k_fc<<<288, 64, 0, stream>>>(dd2, nullptr, nullptr, 1, dfw, dfb, y,   nullptr, nullptr, 256, NPTS*3);

  // ---- fused chamfer (both directions) + final reduce ----
  k_chamfer_min<<<dim3(NPTS/128, 8), 256, 0, stream>>>(y, 0, x, 1, dmin);
  k_chamfer_min<<<dim3(NPTS/128, 8), 256, 0, stream>>>(x, 1, y, 0, dmin + (size_t)NB*NPTS);
  k_final<<<1, 256, 0, stream>>>(dmin, kl, out, 2*NB*NPTS);
}
